// ConvLayer_34282428956962
// MI455X (gfx1250) — matrix operand order repaired and checked by reading, not compiled, not run on hardware
//
#include <hip/hip_runtime.h>
#include <hip/hip_bf16.h>

// Problem constants (from the reference)
#define NNODES 50000
#define MNBR   16
#define FNODE  128
#define FEDGE  64
#define KDIM   320      // 2*FNODE + FEDGE
#define KSTEPS 10       // 320 / 32
#define APITCH 328      // LDS row pitch in bf16 elements (656 B, 16B aligned)

typedef __bf16 v16bf __attribute__((ext_vector_type(16)));
typedef float  v8f   __attribute__((ext_vector_type(8)));

union Frag {            // 32 bytes: one WMMA 16-bit A/B operand per lane
    v16bf v;
    uint4 u[2];
};

// Precise RNE (used for weights only)
static __device__ inline unsigned short f32_to_bf16_rne(float f) {
    unsigned int u = __builtin_bit_cast(unsigned int, f);
    u += 0x7FFFu + ((u >> 16) & 1u);
    return (unsigned short)(u >> 16);
}

// Fast pack for activations: round-half-up + one v_perm_b32 to gather hi16s.
static __device__ inline unsigned int pack_bf16x2(float lo, float hi) {
    unsigned int a = __builtin_bit_cast(unsigned int, lo) + 0x8000u;
    unsigned int b = __builtin_bit_cast(unsigned int, hi) + 0x8000u;
    // dst = { b[31:16], a[31:16] } : sel bytes {7,6,3,2} over {S0=b, S1=a}
    return __builtin_amdgcn_perm(b, a, 0x07060302u);
}
static __device__ inline uint4 pack_bf16x8(float4 a, float4 b) {
    uint4 p;
    p.x = pack_bf16x2(a.x, a.y); p.y = pack_bf16x2(a.z, a.w);
    p.z = pack_bf16x2(b.x, b.y); p.w = pack_bf16x2(b.z, b.w);
    return p;
}
static __device__ inline float softplus_f(float x) {
    return fmaxf(x, 0.f) + __logf(1.f + __expf(-fabsf(x)));
}

// ---- prep: convert W (256 x 320, row-major f32) to bf16 in workspace ----
__global__ void w_to_bf16_kernel(const float* __restrict__ W,
                                 unsigned short* __restrict__ Wbf, int n) {
    int i = blockIdx.x * blockDim.x + threadIdx.x;
    if (i < n) Wbf[i] = f32_to_bf16_rne(W[i]);
}

// ---- main fused conv kernel ----
// Block: 256 threads = 8 waves. Wave w computes output columns [16w, 16w+16):
// filter tile = W rows [16w,16w+16), core tile = W rows [128+16w, 128+16w+16).
// Pipeline per node: prefetch next gather -> 20 WMMAs -> epilogue ->
// pack+store next LDS tile -> single barrier.
__global__ __launch_bounds__(256, 1) void cgcnn_conv_kernel(
    const float* __restrict__ nodeF,   // (N, 128)
    const float* __restrict__ edgeF,   // (N, 16, 64)
    const unsigned short* __restrict__ Wbf,  // (256, 320) bf16
    const float* __restrict__ bias,    // (256)
    const float* __restrict__ alphap,  // scalar
    const int*   __restrict__ idx,     // (N, 16)
    float* __restrict__ out)           // (N, 128)
{
    __shared__ unsigned short ldsA[2][MNBR * APITCH]; // double-buffered 16x320 tile
    __shared__ float ldsMask[2][MNBR];

    const int tid   = threadIdx.x;
    const int wave  = tid >> 5;
    const int lane  = tid & 31;
    const int lhalf = lane >> 4;   // 0: lanes 0-15, 1: lanes 16-31
    const int l15   = lane & 15;
    const int o0    = wave * 16;   // this wave's output-column tile
    const int row   = tid >> 4;    // builder role: neighbor row 0..15
    const int t16   = tid & 15;    // builder role: 16 threads per row

    const float alpha = *alphap;
    const float bF = bias[o0 + l15];
    const float bC = bias[FNODE + o0 + l15];

    // ---- register-resident B fragments (filter + core tiles, 10 k-steps).
    // B k order used here (same as A): lane = column N, lanes 0-15 hold K=+0..15,
    // lanes 16-31 hold K=+16..31 -> contiguous 32B per lane from W row-major.
    Frag bFr[KSTEPS], bCr[KSTEPS];
    {
        const unsigned short* rowF = Wbf + (size_t)(o0 + l15) * KDIM;
        const unsigned short* rowC = Wbf + (size_t)(FNODE + o0 + l15) * KDIM;
        const int kbase = lhalf * 16;
#pragma unroll
        for (int t = 0; t < KSTEPS; ++t) {
            const uint4* pF = (const uint4*)(rowF + t * 32 + kbase);
            const uint4* pC = (const uint4*)(rowC + t * 32 + kbase);
            bFr[t].u[0] = pF[0]; bFr[t].u[1] = pF[1];
            bCr[t].u[0] = pC[0]; bCr[t].u[1] = pC[1];
        }
    }

    int node = blockIdx.x;

    // ---- prologue: build tile for first node into buffer 0
    {
        const int nb  = idx[node * MNBR + row];
        const int gnb = (nb < 0) ? node : nb;
        const float* selfRow = nodeF + (size_t)node * FNODE;
        const float* nbrRow  = nodeF + (size_t)gnb * FNODE;
        const float* edgRow  = edgeF + ((size_t)node * MNBR + row) * FEDGE;
        float4 s0 = ((const float4*)(selfRow + t16 * 8))[0];
        float4 s1 = ((const float4*)(selfRow + t16 * 8))[1];
        float4 n0 = ((const float4*)(nbrRow + t16 * 8))[0];
        float4 n1 = ((const float4*)(nbrRow + t16 * 8))[1];
        float4 e0 = ((const float4*)(edgRow + t16 * 4))[0];
        unsigned short* dst = ldsA[0] + row * APITCH;
        *(uint4*)(dst + t16 * 8)         = pack_bf16x8(s0, s1);
        *(uint4*)(dst + FNODE + t16 * 8) = pack_bf16x8(n0, n1);
        uint2 pe = { pack_bf16x2(e0.x, e0.y), pack_bf16x2(e0.z, e0.w) };
        *(uint2*)(dst + 2 * FNODE + t16 * 4) = pe;
        if (t16 == 0) ldsMask[0][row] = (nb >= 0) ? 1.0f : 0.0f;
    }
    __syncthreads();

    int p = 0;
    for (; node < NNODES; node += gridDim.x) {
        const int  nextNode = node + gridDim.x;
        const bool hasNext  = nextNode < NNODES;   // block-uniform

        // ---- 1) issue next node's gather into registers (overlaps WMMAs)
        float4 s0, s1, n0, n1, e0;
        float  maskv = 0.f;
        if (hasNext) {
            const int nb  = idx[nextNode * MNBR + row];
            maskv = (nb >= 0) ? 1.0f : 0.0f;
            const int gnb = (nb < 0) ? nextNode : nb;
            const float* selfRow = nodeF + (size_t)nextNode * FNODE;
            const float* nbrRow  = nodeF + (size_t)gnb * FNODE;
            const float* edgRow  = edgeF + ((size_t)nextNode * MNBR + row) * FEDGE;
            s0 = ((const float4*)(selfRow + t16 * 8))[0];
            s1 = ((const float4*)(selfRow + t16 * 8))[1];
            n0 = ((const float4*)(nbrRow + t16 * 8))[0];
            n1 = ((const float4*)(nbrRow + t16 * 8))[1];
            e0 = ((const float4*)(edgRow + t16 * 4))[0];
        }

        // ---- 2) 20 WMMAs on current tile, A-fragment pipelined one k-step ahead
        v8f accF, accC;                 // bias folded into C operand
#pragma unroll
        for (int r = 0; r < 8; ++r) { accF[r] = bF; accC[r] = bC; }
        {
            // A k order used here (same as B): lane row = l15; lanes 0-15 take K=+0..15, lanes 16-31 take K=+16..31
            const unsigned short* arow = ldsA[p] + l15 * APITCH + lhalf * 16;
            Frag a, an;
            { const uint4* p0 = (const uint4*)arow; an.u[0] = p0[0]; an.u[1] = p0[1]; }
#pragma unroll
            for (int t = 0; t < KSTEPS; ++t) {
                a = an;
                if (t + 1 < KSTEPS) {
                    const uint4* pn = (const uint4*)(arow + (t + 1) * 32);
                    an.u[0] = pn[0]; an.u[1] = pn[1];
                }
                accF = __builtin_amdgcn_wmma_f32_16x16x32_bf16(
                    false, a.v, false, bFr[t].v, (short)0, accF, false, false);
                accC = __builtin_amdgcn_wmma_f32_16x16x32_bf16(
                    false, a.v, false, bCr[t].v, (short)0, accC, false, false);
            }
        }

        // ---- 3) fused gate * core, masked neighbor reduction (in registers)
        // acc[r]: lane -> neighbor M = r + lhalf*8, column N = o0 + l15
        float s = 0.f;
#pragma unroll
        for (int r = 0; r < 8; ++r) {
            float g = 1.f / (1.f + __expf(-accF[r]));
            float c = softplus_f(accC[r]);
            s += g * c * ldsMask[p][lhalf * 8 + r];
        }
        s += __shfl_xor(s, 16, 32);   // combine neighbor halves across lane groups
        if (lane < 16) {
            const int col = o0 + lane;
            float x = alpha * nodeF[(size_t)node * FNODE + col] + s;
            out[(size_t)node * FNODE + col] = softplus_f(x);
        }

        // ---- 4) convert + store next tile into the other LDS buffer
        if (hasNext) {
            unsigned short* dst = ldsA[p ^ 1] + row * APITCH;
            *(uint4*)(dst + t16 * 8)         = pack_bf16x8(s0, s1);
            *(uint4*)(dst + FNODE + t16 * 8) = pack_bf16x8(n0, n1);
            uint2 pe = { pack_bf16x2(e0.x, e0.y), pack_bf16x2(e0.z, e0.w) };
            *(uint2*)(dst + 2 * FNODE + t16 * 4) = pe;
            if (t16 == 0) ldsMask[p ^ 1][row] = maskv;
        }

        __syncthreads();   // next tile ready; all readers of old tile done
        p ^= 1;
    }
}

extern "C" void kernel_launch(void* const* d_in, const int* in_sizes, int n_in,
                              void* d_out, int out_size, void* d_ws, size_t ws_size,
                              hipStream_t stream) {
    const float* nodeF  = (const float*)d_in[0];   // (N, 128)
    const float* edgeF  = (const float*)d_in[1];   // (N, 16, 64)
    const float* W      = (const float*)d_in[2];   // (256, 320)
    const float* bias   = (const float*)d_in[3];   // (256)
    const float* alphap = (const float*)d_in[4];   // scalar
    const int*   idx    = (const int*)d_in[5];     // (N, 16)
    float* out = (float*)d_out;

    unsigned short* Wbf = (unsigned short*)d_ws;   // 256*320 bf16 = 160 KB

    const int wElems = 2 * FNODE * KDIM;           // 81920
    w_to_bf16_kernel<<<(wElems + 255) / 256, 256, 0, stream>>>(W, Wbf, wElems);

    cgcnn_conv_kernel<<<1024, 256, 0, stream>>>(nodeF, edgeF, Wbf, bias,
                                                alphap, idx, out);
}
